// GATEncoder_55654186222288
// MI455X (gfx1250) — compile-verified, matrix operand order checked by reading, not run on hardware
//
#include <hip/hip_runtime.h>

#define N_NODES 30000
#define E_EDGES 240000
#define ETOT    (E_EDGES + N_NODES)   // 270000 (self-loops appended)
#define IN_CH   256
#define HID     128
#define HEADS   4
#define LAT     128
#define FWIDE   512                    // H*HID == H*LAT == 512
#define NEG_SLOPE 0.2f

typedef __attribute__((ext_vector_type(16))) _Float16 v16h;
typedef __attribute__((ext_vector_type(8)))  _Float16 v8h;
typedef __attribute__((ext_vector_type(8)))  float    v8f;
typedef __attribute__((ext_vector_type(4)))  float    v4f;

// ---------- helpers ----------
__device__ __forceinline__ int f2ord(float f) {
    int i = __float_as_int(f);
    return i >= 0 ? i : (i ^ 0x7fffffff);
}
__device__ __forceinline__ float ord2f(int i) {
    return __int_as_float(i >= 0 ? i : (i ^ 0x7fffffff));
}

// ---------- conversion / transpose ----------
__global__ void f32_to_f16_kernel(const float* __restrict__ src,
                                  _Float16* __restrict__ dst, int n) {
    int i = blockIdx.x * blockDim.x + threadIdx.x;
    if (i < n) dst[i] = (_Float16)src[i];
}

// W: [K][Nc] row-major f32  ->  Wt: [Nc][K] row-major f16
__global__ void transpose_f16_kernel(const float* __restrict__ W,
                                     _Float16* __restrict__ Wt, int K, int Nc) {
    int i = blockIdx.x * blockDim.x + threadIdx.x;
    if (i < K * Nc) {
        int k = i / Nc, n = i - k * Nc;
        Wt[(size_t)n * K + k] = (_Float16)W[i];
    }
}

// ---------- WMMA GEMM: C[M][512] = A[M][K](f16) * Bt[512][K](f16)^T ----------
// grid.x = M/16 ; block = 256 threads (8 waves); wave w -> cols [64w, 64w+64)
// A stripe (16 x K) staged once into LDS; B rows streamed from global (L2-resident).
#define LDS_PAD 8
__global__ __launch_bounds__(256)
void wmma_gemm_kernel(const _Float16* __restrict__ A,
                      const _Float16* __restrict__ Bt,
                      float* __restrict__ C, int K) {
    __shared__ _Float16 Als[16 * (FWIDE + LDS_PAD)];   // max K = 512
    const int stride = K + LDS_PAD;                     // halves; 16B-aligned rows
    const int row0 = blockIdx.x * 16;

    // cooperative stage: 16 rows x K halves, b128 granules (K/8 per row)
    const int cpr = K >> 3;                             // chunks per row
    for (int t = threadIdx.x; t < 16 * cpr; t += 256) {
        int r = t / cpr;
        int c = t - r * cpr;
        *(v8h*)(Als + r * stride + c * 8) =
            *(const v8h*)(A + (size_t)(row0 + r) * K + c * 8);
    }
    __syncthreads();

    const int wave = threadIdx.x >> 5;
    const int lane = threadIdx.x & 31;
    const int col0 = wave * 64;
    const int mr   = lane & 15;          // row (A) / col (B) within tile
    const int kh   = (lane >> 4) * 8;    // K sub-offset per half-wave

    const _Float16* Arow = Als + mr * stride;

    v8f acc[4];
#pragma unroll
    for (int ct = 0; ct < 4; ++ct) acc[ct] = (v8f){0.f,0.f,0.f,0.f,0.f,0.f,0.f,0.f};

    for (int k0 = 0; k0 < K; k0 += 32) {
        v8h alo = *(const v8h*)(Arow + k0 + kh);
        v8h ahi = *(const v8h*)(Arow + k0 + 16 + kh);
        v16h af = __builtin_shufflevector(alo, ahi,
                    0,1,2,3,4,5,6,7,8,9,10,11,12,13,14,15);
#pragma unroll
        for (int ct = 0; ct < 4; ++ct) {
            const _Float16* Brow = Bt + (size_t)(col0 + ct * 16 + mr) * K;
            v8h blo = *(const v8h*)(Brow + k0 + kh);
            v8h bhi = *(const v8h*)(Brow + k0 + 16 + kh);
            v16h bf = __builtin_shufflevector(blo, bhi,
                        0,1,2,3,4,5,6,7,8,9,10,11,12,13,14,15);
            acc[ct] = __builtin_amdgcn_wmma_f32_16x16x32_f16(
                false, af, false, bf, (short)0, acc[ct], false, false);
        }
    }

    const int rbase = row0 + ((lane >> 4) << 3);   // +0 or +8
#pragma unroll
    for (int ct = 0; ct < 4; ++ct) {
        int ccol = col0 + ct * 16 + mr;
#pragma unroll
        for (int r = 0; r < 8; ++r)
            C[(size_t)(rbase + r) * FWIDE + ccol] = acc[ct][r];
    }
}

// ---------- attention logits: als/ald [N][H] ----------
__global__ void attn_logits_kernel(const float* __restrict__ xh,
                                   const float* __restrict__ a_src,
                                   const float* __restrict__ a_dst,
                                   float* __restrict__ als,
                                   float* __restrict__ ald) {
    int i = blockIdx.x * blockDim.x + threadIdx.x;
    if (i >= N_NODES * HEADS) return;
    int n = i >> 2, h = i & 3;
    const float* v  = xh + (size_t)n * FWIDE + h * HID;
    const float* as = a_src + h * HID;
    const float* ad = a_dst + h * HID;
    float ss = 0.f, sd = 0.f;
    for (int c = 0; c < HID; c += 4) {
        v4f xv = *(const v4f*)(v + c);
        v4f sv = *(const v4f*)(as + c);
        v4f dv = *(const v4f*)(ad + c);
#pragma unroll
        for (int t = 0; t < 4; ++t) { ss += xv[t] * sv[t]; sd += xv[t] * dv[t]; }
    }
    als[i] = ss;
    ald[i] = sd;
}

// ---------- init ----------
__global__ void init_mz_kernel(int* __restrict__ m, float* __restrict__ z, int n) {
    int i = blockIdx.x * blockDim.x + threadIdx.x;
    if (i < n) { m[i] = f2ord(-INFINITY); z[i] = 0.f; }
}
__global__ void zero4_kernel(float* __restrict__ p, int nvec4) {
    int i = blockIdx.x * blockDim.x + threadIdx.x;
    if (i < nvec4) ((v4f*)p)[i] = (v4f){0.f, 0.f, 0.f, 0.f};
}

// ---------- edge phase ----------
__global__ void edge_logit_max_kernel(const int* __restrict__ ei,
                                      const float* __restrict__ als,
                                      const float* __restrict__ ald,
                                      float* __restrict__ ebuf,
                                      int* __restrict__ mord) {
    int i = blockIdx.x * blockDim.x + threadIdx.x;
    if (i >= ETOT * HEADS) return;
    int e = i >> 2, h = i & 3;
    int s, d;
    if (e < E_EDGES) { s = ei[e]; d = ei[E_EDGES + e]; } else { s = d = e - E_EDGES; }
    float v = als[s * HEADS + h] + ald[d * HEADS + h];
    v = v > 0.f ? v : NEG_SLOPE * v;
    ebuf[i] = v;
    atomicMax(mord + d * HEADS + h, f2ord(v));
}

__global__ void edge_exp_sum_kernel(const int* __restrict__ ei,
                                    float* __restrict__ ebuf,
                                    const int* __restrict__ mord,
                                    float* __restrict__ z) {
    int i = blockIdx.x * blockDim.x + threadIdx.x;
    if (i >= ETOT * HEADS) return;
    int e = i >> 2, h = i & 3;
    int d;
    if (e < E_EDGES) { d = ei[E_EDGES + e]; } else { d = e - E_EDGES; }
    float m = ord2f(mord[d * HEADS + h]);
    float x = __expf(ebuf[i] - m);
    ebuf[i] = x;
    atomicAdd(z + d * HEADS + h, x);
}

__global__ void edge_alpha_kernel(const int* __restrict__ ei,
                                  float* __restrict__ ebuf,
                                  const float* __restrict__ z) {
    int i = blockIdx.x * blockDim.x + threadIdx.x;
    if (i >= ETOT * HEADS) return;
    int e = i >> 2, h = i & 3;
    int d;
    if (e < E_EDGES) { d = ei[E_EDGES + e]; } else { d = e - E_EDGES; }
    ebuf[i] = ebuf[i] / (z[d * HEADS + h] + 1e-16f);
}

// float4 granules: idx over ETOT * 128
__global__ void aggregate_kernel(const int* __restrict__ ei,
                                 const float* __restrict__ alpha,
                                 const float* __restrict__ xh,
                                 float* __restrict__ agg) {
    const long long total = (long long)ETOT * (FWIDE / 4);
    long long stride = (long long)gridDim.x * blockDim.x;
    for (long long idx = (long long)blockIdx.x * blockDim.x + threadIdx.x;
         idx < total; idx += stride) {
        int e = (int)(idx >> 7);
        int q = (int)(idx & 127);        // float4 index within row
        int h = q >> 5;                  // 32 float4s per head
        int s, d;
        if (e < E_EDGES) { s = ei[e]; d = ei[E_EDGES + e]; } else { s = d = e - E_EDGES; }
        float a = alpha[(size_t)e * HEADS + h];
        v4f xv = *(const v4f*)(xh + (size_t)s * FWIDE + q * 4);
        float* dp = agg + (size_t)d * FWIDE + q * 4;
        atomicAdd(dp + 0, a * xv[0]);
        atomicAdd(dp + 1, a * xv[1]);
        atomicAdd(dp + 2, a * xv[2]);
        atomicAdd(dp + 3, a * xv[3]);
    }
}

// ---------- epilogues ----------
__global__ void bias_relu_f16_kernel(const float* __restrict__ agg,
                                     const float* __restrict__ b,
                                     _Float16* __restrict__ h16) {
    int i = blockIdx.x * blockDim.x + threadIdx.x;
    if (i >= N_NODES * FWIDE) return;
    float v = agg[i] + b[i & 511];
    h16[i] = (_Float16)(v > 0.f ? v : 0.f);
}

__global__ void mean_bias_kernel(const float* __restrict__ agg,
                                 const float* __restrict__ b2,
                                 float* __restrict__ out) {
    int i = blockIdx.x * blockDim.x + threadIdx.x;
    if (i >= N_NODES * LAT) return;
    int n = i >> 7, c = i & 127;
    const float* p = agg + (size_t)n * FWIDE + c;
    out[i] = 0.25f * (p[0] + p[128] + p[256] + p[384]) + b2[c];
}

// =====================================================================
static inline int cdiv(int a, int b) { return (a + b - 1) / b; }

extern "C" void kernel_launch(void* const* d_in, const int* in_sizes, int n_in,
                              void* d_out, int out_size, void* d_ws, size_t ws_size,
                              hipStream_t stream) {
    const float* x       = (const float*)d_in[0];
    const int*   ei      = (const int*)  d_in[1];
    const float* W1      = (const float*)d_in[2];
    const float* a_src1  = (const float*)d_in[3];
    const float* a_dst1  = (const float*)d_in[4];
    const float* b1      = (const float*)d_in[5];
    const float* W2      = (const float*)d_in[6];
    const float* a_src2  = (const float*)d_in[7];
    const float* a_dst2  = (const float*)d_in[8];
    const float* b2      = (const float*)d_in[9];
    float* out = (float*)d_out;

    // ---- workspace carving ----
    char* ws = (char*)d_ws;
    size_t off = 0;
    auto carve = [&](size_t bytes) -> void* {
        void* p = ws + off;
        off = (off + bytes + 255) & ~(size_t)255;
        return p;
    };
    _Float16* x16  = (_Float16*)carve((size_t)N_NODES * IN_CH * 2);
    _Float16* w1t  = (_Float16*)carve((size_t)IN_CH * FWIDE * 2);
    _Float16* w2t  = (_Float16*)carve((size_t)FWIDE * FWIDE * 2);
    float*    xh   = (float*)   carve((size_t)N_NODES * FWIDE * 4);
    float*    agg  = (float*)   carve((size_t)N_NODES * FWIDE * 4);
    _Float16* h16  = (_Float16*)carve((size_t)N_NODES * FWIDE * 2);
    float*    als  = (float*)   carve((size_t)N_NODES * HEADS * 4);
    float*    ald  = (float*)   carve((size_t)N_NODES * HEADS * 4);
    int*      mord = (int*)     carve((size_t)N_NODES * HEADS * 4);
    float*    z    = (float*)   carve((size_t)N_NODES * HEADS * 4);
    float*    ebuf = (float*)   carve((size_t)ETOT * HEADS * 4);

    const int T = 256;
    const int nNH  = N_NODES * HEADS;          // 120000
    const int nEH  = ETOT * HEADS;             // 1080000
    const int nNF  = N_NODES * FWIDE;          // 15360000

    // ---- precision prep ----
    f32_to_f16_kernel<<<cdiv(N_NODES * IN_CH, T), T, 0, stream>>>(x, x16, N_NODES * IN_CH);
    transpose_f16_kernel<<<cdiv(IN_CH * FWIDE, T), T, 0, stream>>>(W1, w1t, IN_CH, FWIDE);
    transpose_f16_kernel<<<cdiv(FWIDE * FWIDE, T), T, 0, stream>>>(W2, w2t, FWIDE, FWIDE);

    // ================= layer 1 =================
    wmma_gemm_kernel<<<N_NODES / 16, T, 0, stream>>>(x16, w1t, xh, IN_CH);
    attn_logits_kernel<<<cdiv(nNH, T), T, 0, stream>>>(xh, a_src1, a_dst1, als, ald);
    init_mz_kernel<<<cdiv(nNH, T), T, 0, stream>>>(mord, z, nNH);
    zero4_kernel<<<cdiv(nNF / 4, T), T, 0, stream>>>(agg, nNF / 4);
    edge_logit_max_kernel<<<cdiv(nEH, T), T, 0, stream>>>(ei, als, ald, ebuf, mord);
    edge_exp_sum_kernel<<<cdiv(nEH, T), T, 0, stream>>>(ei, ebuf, mord, z);
    edge_alpha_kernel<<<cdiv(nEH, T), T, 0, stream>>>(ei, ebuf, z);
    aggregate_kernel<<<4096, T, 0, stream>>>(ei, ebuf, xh, agg);
    bias_relu_f16_kernel<<<cdiv(nNF, T), T, 0, stream>>>(agg, b1, h16);

    // ================= layer 2 =================
    wmma_gemm_kernel<<<N_NODES / 16, T, 0, stream>>>(h16, w2t, xh, FWIDE);
    attn_logits_kernel<<<cdiv(nNH, T), T, 0, stream>>>(xh, a_src2, a_dst2, als, ald);
    init_mz_kernel<<<cdiv(nNH, T), T, 0, stream>>>(mord, z, nNH);
    zero4_kernel<<<cdiv(nNF / 4, T), T, 0, stream>>>(agg, nNF / 4);
    edge_logit_max_kernel<<<cdiv(nEH, T), T, 0, stream>>>(ei, als, ald, ebuf, mord);
    edge_exp_sum_kernel<<<cdiv(nEH, T), T, 0, stream>>>(ei, ebuf, mord, z);
    edge_alpha_kernel<<<cdiv(nEH, T), T, 0, stream>>>(ei, ebuf, z);
    aggregate_kernel<<<4096, T, 0, stream>>>(ei, ebuf, xh, agg);
    mean_bias_kernel<<<cdiv(N_NODES * LAT, T), T, 0, stream>>>(agg, b2, out);
}
